// MaskedViT3D_85529978733181
// MI455X (gfx1250) — compile-verified, matrix operand order checked by reading, not run on hardware
//
#include <hip/hip_runtime.h>
#include <stdint.h>

// ---------------- problem dims ----------------
#define B_    8
#define S_    48
#define ND_   12
#define N_    1728
#define T_TOK 1729
#define D_    192
#define NH_   4
#define L_    4
#define NC_   2
#define DH_   48
#define HID_  768
#define MROWS (B_ * T_TOK)   // 13832 token rows

// ---------------- WMMA plumbing ----------------
typedef __attribute__((ext_vector_type(16))) __bf16 v16bf;
typedef __attribute__((ext_vector_type(8)))  float  v8f;

union FragU {
  v16bf v;
  unsigned short s[16];
  uint4 q[2];
};

__device__ __forceinline__ unsigned short f2bf(float f) {
  union { float f; unsigned u; } x{f};
  unsigned r = x.u + 0x7FFFu + ((x.u >> 16) & 1u);   // round-to-nearest-even
  return (unsigned short)(r >> 16);
}

__device__ __forceinline__ v8f wmma_bf16(v16bf a, v16bf b, v8f c) {
  // (neg_a, A, neg_b, B, c_mod, C, reuse_a, reuse_b)
  return __builtin_amdgcn_wmma_f32_16x16x32_bf16(false, a, false, b, (short)0, c,
                                                 false, false);
}

// ---------------- fp32 -> bf16 convert ----------------
__global__ void cvt_kernel(const float* __restrict__ in, unsigned short* __restrict__ out, int n) {
  int i = blockIdx.x * blockDim.x + threadIdx.x;
  if (i < n) out[i] = f2bf(in[i]);
}

// ---------------- patch gather: x[b,1,48^3] -> patches[b*N, 64] bf16 ----------------
__global__ void gather_patches_kernel(const float* __restrict__ x,
                                      unsigned short* __restrict__ patches, int total) {
  int idx = blockIdx.x * blockDim.x + threadIdx.x;
  if (idx >= total) return;
  int e   = idx & 63;
  int row = idx >> 6;                 // [0, B*N)
  int n   = row % N_;
  int b   = row / N_;
  int pd  = n / (ND_ * ND_);
  int rem = n % (ND_ * ND_);
  int ph  = rem / ND_;
  int pw  = rem % ND_;
  int i = e >> 4, j = (e >> 2) & 3, k = e & 3;
  size_t off = (((size_t)b * S_ + (pd * 4 + i)) * S_ + (ph * 4 + j)) * S_ + (pw * 4 + k);
  patches[idx] = f2bf(x[off]);
}

// ---------------- mask avgpool -> valid flags (1.0/0.0) per key token ----------------
__global__ void valid_kernel(const float* __restrict__ mask, float* __restrict__ valid, int total) {
  int idx = blockIdx.x * blockDim.x + threadIdx.x;
  if (idx >= total) return;
  int n = idx % N_;
  int b = idx / N_;
  int pd  = n / (ND_ * ND_);
  int rem = n % (ND_ * ND_);
  int ph  = rem / ND_;
  int pw  = rem % ND_;
  float s = 0.f;
  for (int i = 0; i < 4; ++i)
    for (int j = 0; j < 4; ++j)
      for (int k = 0; k < 4; ++k) {
        size_t off = (((size_t)b * S_ + (pd * 4 + i)) * S_ + (ph * 4 + j)) * S_ + (pw * 4 + k);
        s += mask[off];
      }
  valid[(size_t)b * T_TOK + 1 + n] = (s * (1.f / 64.f) > 0.5f) ? 1.f : 0.f;
  if (n == 0) valid[(size_t)b * T_TOK] = 1.f;   // cls always valid
}

// ---------------- tok = concat(cls, emb) + pos (fp32) ----------------
__global__ void assemble_kernel(const float* __restrict__ emb, const float* __restrict__ cls,
                                const float* __restrict__ pos, float* __restrict__ tok, int total) {
  int idx = blockIdx.x * blockDim.x + threadIdx.x;
  if (idx >= total) return;
  int d = idx % D_;
  int t = (idx / D_) % T_TOK;
  int b = idx / (D_ * T_TOK);
  float v = (t == 0) ? cls[d] : emb[((size_t)b * N_ + (t - 1)) * D_ + d];
  tok[idx] = v + pos[(size_t)t * D_ + d];
}

// ---------------- LayerNorm: fp32 in -> bf16 out, one wave per row ----------------
__global__ __launch_bounds__(256) void ln_kernel(const float* __restrict__ x,
                                                 const float* __restrict__ w,
                                                 const float* __restrict__ b,
                                                 unsigned short* __restrict__ y, int M, int Dd) {
  int row  = blockIdx.x * 8 + (threadIdx.x >> 5);
  int lane = threadIdx.x & 31;
  if (row >= M) return;
  const float* xr = x + (size_t)row * Dd;
  float s = 0.f, s2 = 0.f;
  for (int i = lane; i < Dd; i += 32) { float v = xr[i]; s += v; s2 += v * v; }
  for (int off = 16; off > 0; off >>= 1) {
    s  += __shfl_xor(s, off, 32);
    s2 += __shfl_xor(s2, off, 32);
  }
  float mean = s / Dd;
  float rstd = rsqrtf(s2 / Dd - mean * mean + 1e-5f);
  unsigned short* yr = y + (size_t)row * Dd;
  for (int i = lane; i < Dd; i += 32)
    yr[i] = f2bf((xr[i] - mean) * rstd * w[i] + b[i]);
}

// ---------------- WMMA GEMM: out = act(A[MxK] * W[NxK]^T + bias) (+resid) ----------------
// block = 128 threads = 4 waves, 64x64 tile, each wave a 32x32 subtile (2x2 fragments)
__global__ __launch_bounds__(128) void gemm_bf16_kernel(
    const unsigned short* __restrict__ A, const unsigned short* __restrict__ W,
    const float* __restrict__ bias, const float* __restrict__ resid,
    float* __restrict__ outF, unsigned short* __restrict__ outB,
    int M, int Ncols, int K, int act) {
  const int wid  = threadIdx.x >> 5;
  const int lane = threadIdx.x & 31;
  const int lh   = lane >> 4, lm = lane & 15;
  const int m0 = blockIdx.x * 64 + (wid >> 1) * 32;
  const int n0 = blockIdx.y * 64 + (wid & 1) * 32;

  v8f c00 = {}, c01 = {}, c10 = {}, c11 = {};
  int ra0 = m0 + lm;      if (ra0 > M - 1) ra0 = M - 1;
  int ra1 = m0 + 16 + lm; if (ra1 > M - 1) ra1 = M - 1;
  const unsigned short* a0p = A + (size_t)ra0 * K + 8 * lh;
  const unsigned short* a1p = A + (size_t)ra1 * K + 8 * lh;
  const unsigned short* b0p = W + (size_t)(n0 + lm) * K + 8 * lh;
  const unsigned short* b1p = W + (size_t)(n0 + 16 + lm) * K + 8 * lh;

  for (int k0 = 0; k0 < K; k0 += 32) {
    FragU a0, a1, b0, b1;
    a0.q[0] = *(const uint4*)(a0p + k0); a0.q[1] = *(const uint4*)(a0p + k0 + 16);
    a1.q[0] = *(const uint4*)(a1p + k0); a1.q[1] = *(const uint4*)(a1p + k0 + 16);
    b0.q[0] = *(const uint4*)(b0p + k0); b0.q[1] = *(const uint4*)(b0p + k0 + 16);
    b1.q[0] = *(const uint4*)(b1p + k0); b1.q[1] = *(const uint4*)(b1p + k0 + 16);
    c00 = wmma_bf16(a0.v, b0.v, c00);
    c01 = wmma_bf16(a0.v, b1.v, c01);
    c10 = wmma_bf16(a1.v, b0.v, c10);
    c11 = wmma_bf16(a1.v, b1.v, c11);
  }

  for (int mi = 0; mi < 2; ++mi)
    for (int ni = 0; ni < 2; ++ni) {
      v8f c = (mi == 0) ? ((ni == 0) ? c00 : c01) : ((ni == 0) ? c10 : c11);
      int col  = n0 + ni * 16 + lm;
      float bv = bias ? bias[col] : 0.f;
      for (int r = 0; r < 8; ++r) {
        int row = m0 + mi * 16 + 8 * lh + r;
        if (row < M) {
          float v = c[r] + bv;
          if (act == 1) v = 0.5f * v * (1.0f + erff(v * 0.70710678118654752f));
          size_t idx = (size_t)row * Ncols + col;
          if (resid) v += resid[idx];
          if (outF) outF[idx] = v;
          if (outB) outB[idx] = f2bf(v);
        }
      }
    }
}

// ---------------- flash attention: per (b, h, 64-query block), 4 waves x 16 queries ----------------
// qkv bf16 [B*T, 576]: q = [h*48 .. ], k = [192 + h*48 ..], v = [384 + h*48 ..]
__global__ __launch_bounds__(128) void attn_kernel(const unsigned short* __restrict__ qkv,
                                                   const float* __restrict__ valid,
                                                   unsigned short* __restrict__ attnOut) {
  __shared__ unsigned short pl[4][16][32];   // per-wave P tile for A-fragment transpose

  const int h = blockIdx.y, b = blockIdx.z;
  const int wid = threadIdx.x >> 5, lane = threadIdx.x & 31;
  const int lh = lane >> 4, lm = lane & 15;
  const int q0 = blockIdx.x * 64 + wid * 16;

  const unsigned short* base = qkv + (size_t)b * T_TOK * 576;
  const float* vld = valid + (size_t)b * T_TOK;

  // Q fragments (DH=48 zero-padded to K=64, two 16x32 A fragments)
  int qr = q0 + lm; if (qr > T_TOK - 1) qr = T_TOK - 1;
  const unsigned short* qp = base + (size_t)qr * 576 + h * DH_;
  FragU aq0, aq1;
  aq0.q[0] = *(const uint4*)(qp + 8 * lh);
  aq0.q[1] = *(const uint4*)(qp + 16 + 8 * lh);
  aq1.q[0] = *(const uint4*)(qp + 32 + 8 * lh);
  #pragma unroll
  for (int i = 8; i < 16; ++i) aq1.s[i] = 0;   // d = 48..63 pad

  float m[8], l[8];
  v8f o0 = {}, o1 = {}, o2 = {};
  #pragma unroll
  for (int r = 0; r < 8; ++r) { m[r] = -1e30f; l[r] = 0.f; }
  const float sc = 0.14433756729740643f;  // 1/sqrt(48)

  for (int kb = 0; kb < T_TOK; kb += 32) {
    float s0[8], s1[8];
    #pragma unroll
    for (int half = 0; half < 2; ++half) {
      int kidx = kb + half * 16 + lm;
      int kr = (kidx < T_TOK) ? kidx : T_TOK - 1;
      const unsigned short* kp = base + (size_t)kr * 576 + 192 + h * DH_;
      FragU bk0, bk1;
      bk0.q[0] = *(const uint4*)(kp + 8 * lh);
      bk0.q[1] = *(const uint4*)(kp + 16 + 8 * lh);
      bk1.q[0] = *(const uint4*)(kp + 32 + 8 * lh);
      #pragma unroll
      for (int i = 8; i < 16; ++i) bk1.s[i] = 0;
      v8f acc = {};
      acc = wmma_bf16(aq1.v, bk1.v, acc);
      acc = wmma_bf16(aq0.v, bk0.v, acc);
      float kv = (kidx < T_TOK) ? vld[kidx] : 0.f;   // key-padding mask per column (= lane)
      float* s = half ? s1 : s0;
      #pragma unroll
      for (int r = 0; r < 8; ++r) s[r] = (kv > 0.5f) ? acc[r] * sc : -1e9f;
    }

    // online softmax: each row lives in one 16-lane half (row = 8*lh + r)
    #pragma unroll
    for (int r = 0; r < 8; ++r) {
      float mx = fmaxf(s0[r], s1[r]);
      for (int off = 1; off < 16; off <<= 1) mx = fmaxf(mx, __shfl_xor(mx, off, 32));
      float mn  = fmaxf(m[r], mx);
      float fac = __expf(m[r] - mn);
      float p0  = __expf(s0[r] - mn);
      float p1  = __expf(s1[r] - mn);
      float rs  = p0 + p1;
      for (int off = 1; off < 16; off <<= 1) rs += __shfl_xor(rs, off, 32);
      m[r] = mn;
      l[r] = l[r] * fac + rs;
      o0[r] *= fac; o1[r] *= fac; o2[r] *= fac;
      pl[wid][8 * lh + r][lm]      = f2bf(p0);
      pl[wid][8 * lh + r][16 + lm] = f2bf(p1);
    }
    __syncthreads();

    // P as 16x32 A fragment (transpose through LDS)
    FragU ap;
    #pragma unroll
    for (int i = 0; i < 16; ++i) {
      int k = (i < 8) ? (8 * lh + i) : (16 + 8 * lh + (i - 8));
      ap.s[i] = pl[wid][lm][k];
    }
    // O += P(16x32) @ V(32x48), three 16-wide column groups
    #pragma unroll
    for (int f = 0; f < 3; ++f) {
      FragU bv;
      #pragma unroll
      for (int i = 0; i < 16; ++i) {
        int k = (i < 8) ? (8 * lh + i) : (16 + 8 * lh + (i - 8));
        int kr = kb + k; if (kr > T_TOK - 1) kr = T_TOK - 1;
        bv.s[i] = base[(size_t)kr * 576 + 384 + h * DH_ + f * 16 + lm];
      }
      if (f == 0)      o0 = wmma_bf16(ap.v, bv.v, o0);
      else if (f == 1) o1 = wmma_bf16(ap.v, bv.v, o1);
      else             o2 = wmma_bf16(ap.v, bv.v, o2);
    }
    __syncthreads();
  }

  #pragma unroll
  for (int f = 0; f < 3; ++f)
    for (int r = 0; r < 8; ++r) {
      int row = q0 + 8 * lh + r;
      if (row < T_TOK) {
        float ov = (f == 0) ? o0[r] : (f == 1) ? o1[r] : o2[r];
        attnOut[((size_t)b * T_TOK + row) * D_ + h * DH_ + f * 16 + lm] = f2bf(ov / l[r]);
      }
    }
}

// ---------------- final LN on cls rows + classifier head ----------------
__global__ void head_kernel(const float* __restrict__ tok, const float* __restrict__ lnw,
                            const float* __restrict__ lnb, const float* __restrict__ hw,
                            const float* __restrict__ hb, float* __restrict__ out) {
  int idx = threadIdx.x;
  if (idx >= B_ * NC_) return;
  int b = idx / NC_, c = idx % NC_;
  const float* xr = tok + (size_t)b * T_TOK * D_;   // cls row (t==0)
  float s = 0.f, s2 = 0.f;
  for (int d = 0; d < D_; ++d) { float v = xr[d]; s += v; s2 += v * v; }
  float mean = s / D_;
  float rstd = rsqrtf(s2 / D_ - mean * mean + 1e-5f);
  float acc = hb[c];
  for (int d = 0; d < D_; ++d)
    acc += ((xr[d] - mean) * rstd * lnw[d] + lnb[d]) * hw[c * D_ + d];
  out[idx] = acc;
}

// ---------------- host orchestration ----------------
extern "C" void kernel_launch(void* const* d_in, const int* in_sizes, int n_in,
                              void* d_out, int out_size, void* d_ws, size_t ws_size,
                              hipStream_t stream) {
  const float* x       = (const float*)d_in[0];
  const float* mask    = (const float*)d_in[1];
  const float* conv_w  = (const float*)d_in[2];
  const float* conv_b  = (const float*)d_in[3];
  const float* cls_tok = (const float*)d_in[4];
  const float* pos     = (const float*)d_in[5];
  const float* ln1_w   = (const float*)d_in[6];
  const float* ln1_b   = (const float*)d_in[7];
  const float* qkv_w   = (const float*)d_in[8];
  const float* qkv_b   = (const float*)d_in[9];
  const float* out_w   = (const float*)d_in[10];
  const float* out_b   = (const float*)d_in[11];
  const float* ln2_w   = (const float*)d_in[12];
  const float* ln2_b   = (const float*)d_in[13];
  const float* mlp_w1  = (const float*)d_in[14];
  const float* mlp_b1  = (const float*)d_in[15];
  const float* mlp_w2  = (const float*)d_in[16];
  const float* mlp_b2  = (const float*)d_in[17];
  const float* lnf_w   = (const float*)d_in[18];
  const float* lnf_b   = (const float*)d_in[19];
  const float* head_w  = (const float*)d_in[20];
  const float* head_b  = (const float*)d_in[21];
  (void)in_sizes; (void)n_in; (void)out_size; (void)ws_size;

  char* basep = (char*)d_ws;
  size_t off = 0;
  auto take = [&](size_t bytes) -> void* {
    void* p = basep + off;
    off += (bytes + 255) & ~(size_t)255;
    return p;
  };
  float*          tok   = (float*)take((size_t)MROWS * D_ * 4);
  unsigned short* ybf   = (unsigned short*)take((size_t)MROWS * D_ * 2);
  unsigned short* qkvbf = (unsigned short*)take((size_t)MROWS * 3 * D_ * 2);
  unsigned short* attbf = (unsigned short*)take((size_t)MROWS * D_ * 2);
  unsigned short* hidbf = (unsigned short*)take((size_t)MROWS * HID_ * 2);
  float*          emb   = (float*)take((size_t)B_ * N_ * D_ * 4);
  unsigned short* patbf = (unsigned short*)take((size_t)B_ * N_ * 64 * 2);
  float*          valid = (float*)take((size_t)MROWS * 4);
  unsigned short* convw_bf = (unsigned short*)take((size_t)D_ * 64 * 2);
  unsigned short* qkvw_bf  = (unsigned short*)take((size_t)L_ * 3 * D_ * D_ * 2);
  unsigned short* outw_bf  = (unsigned short*)take((size_t)L_ * D_ * D_ * 2);
  unsigned short* m1w_bf   = (unsigned short*)take((size_t)L_ * HID_ * D_ * 2);
  unsigned short* m2w_bf   = (unsigned short*)take((size_t)L_ * D_ * HID_ * 2);

  auto cvt = [&](const float* src, unsigned short* dst, int n) {
    cvt_kernel<<<(n + 255) / 256, 256, 0, stream>>>(src, dst, n);
  };
  cvt(conv_w, convw_bf, D_ * 64);
  cvt(qkv_w,  qkvw_bf,  L_ * 3 * D_ * D_);
  cvt(out_w,  outw_bf,  L_ * D_ * D_);
  cvt(mlp_w1, m1w_bf,   L_ * HID_ * D_);
  cvt(mlp_w2, m2w_bf,   L_ * D_ * HID_);

  { int n = B_ * N_ * 64; gather_patches_kernel<<<(n + 255) / 256, 256, 0, stream>>>(x, patbf, n); }
  { int n = B_ * N_;      valid_kernel<<<(n + 255) / 256, 256, 0, stream>>>(mask, valid, n); }

  auto gemm = [&](const unsigned short* A, const unsigned short* W, const float* bias,
                  const float* resid, float* oF, unsigned short* oB,
                  int M, int Nc, int K, int act) {
    dim3 g((M + 63) / 64, Nc / 64);
    gemm_bf16_kernel<<<g, 128, 0, stream>>>(A, W, bias, resid, oF, oB, M, Nc, K, act);
  };

  // patch embed: [B*N,64] @ conv_w[192,64]^T + conv_b -> emb fp32
  gemm(patbf, convw_bf, conv_b, nullptr, emb, nullptr, B_ * N_, D_, 64, 0);
  { int n = MROWS * D_; assemble_kernel<<<(n + 255) / 256, 256, 0, stream>>>(emb, cls_tok, pos, tok, n); }

  for (int l = 0; l < L_; ++l) {
    ln_kernel<<<(MROWS + 7) / 8, 256, 0, stream>>>(tok, ln1_w + l * D_, ln1_b + l * D_, ybf, MROWS, D_);
    gemm(ybf, qkvw_bf + (size_t)l * 3 * D_ * D_, qkv_b + l * 3 * D_,
         nullptr, nullptr, qkvbf, MROWS, 3 * D_, D_, 0);
    attn_kernel<<<dim3((T_TOK + 63) / 64, NH_, B_), 128, 0, stream>>>(qkvbf, valid, attbf);
    gemm(attbf, outw_bf + (size_t)l * D_ * D_, out_b + l * D_,
         tok, tok, nullptr, MROWS, D_, D_, 0);                       // += residual, in-place fp32
    ln_kernel<<<(MROWS + 7) / 8, 256, 0, stream>>>(tok, ln2_w + l * D_, ln2_b + l * D_, ybf, MROWS, D_);
    gemm(ybf, m1w_bf + (size_t)l * HID_ * D_, mlp_b1 + l * HID_,
         nullptr, nullptr, hidbf, MROWS, HID_, D_, 1);               // exact GELU epilogue
    gemm(hidbf, m2w_bf + (size_t)l * D_ * HID_, mlp_b2 + l * D_,
         tok, tok, nullptr, MROWS, D_, HID_, 0);                     // += residual
  }

  head_kernel<<<1, 64, 0, stream>>>(tok, lnf_w, lnf_b, head_w, head_b, (float*)d_out);
}
